// MultiHeadQuantileNBEATS_52089363366111
// MI455X (gfx1250) — matrix operand order repaired and checked by reading, not compiled, not run on hardware
//
#include <hip/hip_runtime.h>

// ---------------------------------------------------------------------------
// Problem constants (from reference)
// ---------------------------------------------------------------------------
#define BATCH 8192
#define TDIM  512
#define DDIM  512
#define H1DIM 256
#define H2DIM 128
#define HORD  96
#define QF    7
#define QT    32

typedef __attribute__((ext_vector_type(16))) __bf16 v16bf;
typedef __attribute__((ext_vector_type(8)))  __bf16 v8bf;
typedef __attribute__((ext_vector_type(8)))  float  v8f;

// ---------------------------------------------------------------------------
// f32 -> bf16 elementwise convert (for x)
// ---------------------------------------------------------------------------
__global__ __launch_bounds__(256) void f32_to_bf16_kernel(
    const float* __restrict__ in, __bf16* __restrict__ out, long n) {
  long i = (long)blockIdx.x * blockDim.x + threadIdx.x;
  if (i < n) out[i] = (__bf16)in[i];
}

// ---------------------------------------------------------------------------
// f32 [Q,K,N] -> bf16 [Q,N,K] transpose-convert (weights; tiny)
// ---------------------------------------------------------------------------
__global__ __launch_bounds__(256) void transpose_to_bf16_kernel(
    const float* __restrict__ in, __bf16* __restrict__ out, int K, int N) {
  int q   = blockIdx.y;
  long i  = (long)blockIdx.x * blockDim.x + threadIdx.x;
  long KN = (long)K * N;
  if (i < KN) {
    int k = (int)(i / N);
    int n = (int)(i % N);
    out[(size_t)q * KN + (size_t)n * K + k] = (__bf16)in[(size_t)q * KN + i];
  }
}

// ---------------------------------------------------------------------------
// Generic batched GEMM with N register blocking:
//   each wave computes NB adjacent 16x16 C tiles (16 x 16*NB), reusing the A
//   fragment NB times with NB independent WMMA accumulator chains.
//   Per K-step, ALL fragment loads are issued first (one clause, distinct
//   registers), then the NB WMMAs consume them -> partial loadcnt waits and
//   overlap with the next (compiler-unrolled) step's load clause.
//   A  : bf16, row m at A + q*a_q_off + m*lda   (lda in elements)
//   Wt : bf16, [Q, N, K] (pre-transposed so B columns are contiguous in K)
// ---------------------------------------------------------------------------
template <int NB, bool RELU, bool OUT_BF16>
__global__ __launch_bounds__(256) void gemm_bf16_wmma_kernel(
    const __bf16* __restrict__ A, int lda, int a_q_off,
    const __bf16* __restrict__ Wt, const float* __restrict__ bias,
    void* __restrict__ out, int out_row_stride, int out_q_off,
    int N, int K, int ntGroups) {
  const int q    = blockIdx.z;
  const int job  = blockIdx.x * 8 + (threadIdx.x >> 5);  // one job per wave
  const int lane = threadIdx.x & 31;
  const int half = lane >> 4;  // which 16-lane half of the wave
  const int r    = lane & 15;  // row (A) / column (B,C) within a tile

  const int mtile = job / ntGroups;
  const int ntg   = job % ntGroups;  // group of NB adjacent N tiles

  const __bf16* Ab = A + (size_t)q * a_q_off + (size_t)(mtile * 16 + r) * lda;
  const __bf16* Bb[NB];
#pragma unroll
  for (int nb = 0; nb < NB; ++nb) {
    int ntile = ntg * NB + nb;
    Bb[nb] = Wt + ((size_t)q * N + (size_t)(ntile * 16 + r)) * K;
  }

  v8f c[NB];
#pragma unroll
  for (int nb = 0; nb < NB; ++nb) c[nb] = (v8f){};

#pragma unroll 2
  for (int k = 0; k < K; k += 32) {
    // ---- load phase: issue ALL fragments for this K-step -----------------
    // A fragment (16x32 bf16), ISA layout: per lane two 8-elem runs at
    //   K = k + half*8 + [0,8)   and   K = k + 16 + half*8 + [0,8)
    v8bf a0 = *(const v8bf*)(Ab + k + half * 8);
    v8bf a1 = *(const v8bf*)(Ab + k + 16 + half * 8);
    // B fragments (32x16 bf16): same layout as A, per lane two 8-elem runs at
    //   K = k + half*8 + [0,8)   and   K = k + 16 + half*8 + [0,8)
    v8bf b0[NB], b1[NB];
#pragma unroll
    for (int nb = 0; nb < NB; ++nb) {
      b0[nb] = *(const v8bf*)(Bb[nb] + k + half * 8);
      b1[nb] = *(const v8bf*)(Bb[nb] + k + 16 + half * 8);
    }

    // ---- compute phase: NB independent accumulator chains ----------------
    v16bf a = __builtin_shufflevector(a0, a1, 0, 1, 2, 3, 4, 5, 6, 7,
                                      8, 9, 10, 11, 12, 13, 14, 15);
#pragma unroll
    for (int nb = 0; nb < NB; ++nb) {
      v16bf b = __builtin_shufflevector(b0[nb], b1[nb], 0, 1, 2, 3, 4, 5, 6, 7,
                                        8, 9, 10, 11, 12, 13, 14, 15);
      c[nb] = __builtin_amdgcn_wmma_f32_16x16x32_bf16(
          /*neg_a=*/false, a, /*neg_b=*/false, b,
          /*c_mod=*/(short)0, c[nb], /*reuse_a=*/false, /*reuse_b=*/false);
    }
  }

  // C layout: elem e of lane -> row m = 8*half + e, col n = lane & 15
#pragma unroll
  for (int nb = 0; nb < NB; ++nb) {
    int ntile = ntg * NB + nb;
    const float bv = bias[(size_t)q * N + ntile * 16 + r];
    size_t obase = (size_t)(mtile * 16 + half * 8) * out_row_stride +
                   (size_t)q * out_q_off + (size_t)ntile * 16 + r;
#pragma unroll
    for (int e = 0; e < 8; ++e) {
      float vv = c[nb][e] + bv;
      if (RELU) vv = fmaxf(vv, 0.0f);
      if (OUT_BF16)
        ((__bf16*)out)[obase + (size_t)e * out_row_stride] = (__bf16)vv;
      else
        ((float*)out)[obase + (size_t)e * out_row_stride] = vv;
    }
  }
}

// ---------------------------------------------------------------------------
// Epilogue: per (b,h): sort 7 quantile preds, lerp to 32 target quantiles
// (bracket against the CONSTANT level grid), sort the 32 results.
// Block = one batch row b, 96 threads = horizon positions.
// ---------------------------------------------------------------------------
__device__ __forceinline__ float pick7(const float* s, int i) {
  float r = s[0];
  r = (i == 1) ? s[1] : r;
  r = (i == 2) ? s[2] : r;
  r = (i == 3) ? s[3] : r;
  r = (i == 4) ? s[4] : r;
  r = (i == 5) ? s[5] : r;
  r = (i == 6) ? s[6] : r;
  return r;
}

__global__ __launch_bounds__(96) void quantile_epilogue_kernel(
    const float* __restrict__ out3,   // [B, QF, HOR]
    const float* __restrict__ qv,     // [B, QT]
    float* __restrict__ dst) {        // [B, HOR, QT]
  const int b = blockIdx.x;
  const int t = threadIdx.x;  // 0..95

  __shared__ float s_w[QT];
  __shared__ int   s_il[QT];
  __shared__ int   s_ih[QT];

  if (t < QT) {
    const float ql[QF] = {0.025f, 0.1f, 0.25f, 0.5f, 0.75f, 0.9f, 0.975f};
    float qq = qv[(size_t)b * QT + t];
    int ih = 0;
#pragma unroll
    for (int i = 0; i < QF; ++i) ih += (ql[i] < qq) ? 1 : 0;  // searchsorted left
    ih = min(max(ih, 1), QF - 1);
    int il = ih - 1;
    float w = (qq - ql[il]) / (ql[ih] - ql[il] + 1e-8f);
    if (qq <= ql[0])      { il = 0;      ih = 0;      w = 0.0f; }
    if (qq >= ql[QF - 1]) { il = QF - 1; ih = QF - 1; w = 0.0f; }
    s_w[t] = w; s_il[t] = il; s_ih[t] = ih;
  }
  __syncthreads();

  // Gather the 7 per-quantile predictions for (b, h=t) and sort ascending.
  float s[QF];
#pragma unroll
  for (int qf = 0; qf < QF; ++qf)
    s[qf] = out3[((size_t)b * QF + qf) * HORD + t];
#pragma unroll
  for (int i = 0; i < QF - 1; ++i)
#pragma unroll
    for (int j = 0; j < QF - 1 - i; ++j) {
      float lo = fminf(s[j], s[j + 1]);
      float hi = fmaxf(s[j], s[j + 1]);
      s[j] = lo; s[j + 1] = hi;
    }

  // Interpolate to the 32 requested quantiles.
  float v[QT];
#pragma unroll
  for (int k = 0; k < QT; ++k) {
    float w  = s_w[k];
    float lo = pick7(s, s_il[k]);
    float hi = pick7(s, s_ih[k]);
    v[k] = (1.0f - w) * lo + w * hi;
  }

  // Bitonic sort the 32 results ascending (fully unrolled -> registers only).
#pragma unroll
  for (int size = 2; size <= QT; size <<= 1) {
#pragma unroll
    for (int stride = size >> 1; stride > 0; stride >>= 1) {
#pragma unroll
      for (int i = 0; i < QT; ++i) {
        int j = i ^ stride;
        if (j > i) {
          bool up = ((i & size) == 0);
          float a = v[i], bb = v[j];
          float lo = fminf(a, bb), hi = fmaxf(a, bb);
          v[i] = up ? lo : hi;
          v[j] = up ? hi : lo;
        }
      }
    }
  }

  size_t obase = ((size_t)b * HORD + t) * QT;
#pragma unroll
  for (int k = 0; k < QT; ++k) dst[obase + k] = v[k];
}

// ---------------------------------------------------------------------------
// Launch
// ---------------------------------------------------------------------------
static inline size_t align256(size_t x) { return (x + 255) & ~(size_t)255; }

extern "C" void kernel_launch(void* const* d_in, const int* in_sizes, int n_in,
                              void* d_out, int out_size, void* d_ws, size_t ws_size,
                              hipStream_t stream) {
  (void)in_sizes; (void)n_in; (void)out_size; (void)ws_size;

  const float* x   = (const float*)d_in[0];  // [B, T]
  const float* qv  = (const float*)d_in[1];  // [B, QT]
  const float* Wbb = (const float*)d_in[2];  // [T, D]
  const float* bbb = (const float*)d_in[3];  // [D]
  const float* W1  = (const float*)d_in[4];  // [QF, D, H1]
  const float* b1  = (const float*)d_in[5];  // [QF, H1]
  const float* W2  = (const float*)d_in[6];  // [QF, H1, H2]
  const float* b2  = (const float*)d_in[7];  // [QF, H2]
  const float* W3  = (const float*)d_in[8];  // [QF, H2, HOR]
  const float* b3  = (const float*)d_in[9];  // [QF, HOR]

  char* ws = (char*)d_ws;
  size_t off = 0;
  __bf16* xb    = (__bf16*)(ws + off); off = align256(off + (size_t)BATCH * TDIM * 2);
  __bf16* wbbT  = (__bf16*)(ws + off); off = align256(off + (size_t)TDIM * DDIM * 2);
  __bf16* w1T   = (__bf16*)(ws + off); off = align256(off + (size_t)QF * H1DIM * DDIM * 2);
  __bf16* w2T   = (__bf16*)(ws + off); off = align256(off + (size_t)QF * H2DIM * H1DIM * 2);
  __bf16* w3T   = (__bf16*)(ws + off); off = align256(off + (size_t)QF * HORD * H2DIM * 2);
  __bf16* feats = (__bf16*)(ws + off); off = align256(off + (size_t)BATCH * DDIM * 2);
  __bf16* h1    = (__bf16*)(ws + off); off = align256(off + (size_t)BATCH * QF * H1DIM * 2);
  __bf16* h2    = (__bf16*)(ws + off); off = align256(off + (size_t)BATCH * QF * H2DIM * 2);
  float*  out3  = (float*)(ws + off);  off = align256(off + (size_t)BATCH * QF * HORD * 4);

  // --- precision conversion / weight transpose -----------------------------
  {
    long n = (long)BATCH * TDIM;
    f32_to_bf16_kernel<<<(unsigned)((n + 255) / 256), 256, 0, stream>>>(x, xb, n);
  }
  transpose_to_bf16_kernel<<<dim3((TDIM * DDIM + 255) / 256, 1), 256, 0, stream>>>(
      Wbb, wbbT, TDIM, DDIM);
  transpose_to_bf16_kernel<<<dim3((DDIM * H1DIM + 255) / 256, QF), 256, 0, stream>>>(
      W1, w1T, DDIM, H1DIM);
  transpose_to_bf16_kernel<<<dim3((H1DIM * H2DIM + 255) / 256, QF), 256, 0, stream>>>(
      W2, w2T, H1DIM, H2DIM);
  transpose_to_bf16_kernel<<<dim3((H2DIM * HORD + 255) / 256, QF), 256, 0, stream>>>(
      W3, w3T, H2DIM, HORD);

  const int MT = BATCH / 16;  // 512 M-tiles

  // --- stage 1: feats = x @ W_bb + b_bb (no relu, bf16 out) ----------------
  // N=512 -> 32 tiles -> 8 groups of 4; jobs = 512*8, 8 waves/block.
  gemm_bf16_wmma_kernel<4, false, true>
      <<<dim3(MT * 8 / 8, 1, 1), 256, 0, stream>>>(
          xb, TDIM, 0, wbbT, bbb, feats, DDIM, 0, DDIM, TDIM, 8);

  // --- stage 2: h1 = relu(feats @ W1[q] + b1[q]) ---------------------------
  // N=256 -> 16 tiles -> 4 groups of 4; jobs = 512*4.
  gemm_bf16_wmma_kernel<4, true, true>
      <<<dim3(MT * 4 / 8, 1, QF), 256, 0, stream>>>(
          feats, DDIM, 0, w1T, b1, h1, QF * H1DIM, H1DIM, H1DIM, DDIM, 4);

  // --- stage 3: h2 = relu(h1[:,q] @ W2[q] + b2[q]) -------------------------
  // N=128 -> 8 tiles -> 2 groups of 4; jobs = 512*2.
  gemm_bf16_wmma_kernel<4, true, true>
      <<<dim3(MT * 2 / 8, 1, QF), 256, 0, stream>>>(
          h1, QF * H1DIM, H1DIM, w2T, b2, h2, QF * H2DIM, H2DIM, H2DIM, H1DIM, 2);

  // --- stage 4: out3 = h2[:,q] @ W3[q] + b3[q] (f32 out) -------------------
  // N=96 -> 6 tiles -> 3 groups of 2; jobs = 512*3.
  gemm_bf16_wmma_kernel<2, false, false>
      <<<dim3(MT * 3 / 8, 1, QF), 256, 0, stream>>>(
          h2, QF * H2DIM, H2DIM, w3T, b3, out3, QF * HORD, HORD, HORD, H2DIM, 3);

  // --- epilogue: sort-7, bracket-lerp, sort-32 -----------------------------
  quantile_epilogue_kernel<<<BATCH, HORD, 0, stream>>>(out3, qv, (float*)d_out);
}
